// MDTA_15934328668212
// MI455X (gfx1250) — matrix operand order repaired and checked by reading, not compiled, not run on hardware
//
#include <hip/hip_runtime.h>
#include <hip/hip_bf16.h>

typedef __attribute__((ext_vector_type(16))) _Float16 v16h;
typedef __attribute__((ext_vector_type(4)))  _Float16 v4h;
typedef __attribute__((ext_vector_type(4)))  float    v4f;
typedef __attribute__((ext_vector_type(8)))  float    v8f;

#define HWSZ 16384L   // 128*128
#define EPSN 1e-12f

union hfrag { v16h v; v4h q[4]; _Float16 e[16]; };

// Convert two runs of 8 contiguous f32 into a v16h fragment (halves 0..7 from p0, 8..15 from p1)
__device__ __forceinline__ v16h cvt16(const float* p0, const float* p1) {
  hfrag r;
  r.q[0] = __builtin_convertvector(*(const v4f*)(p0),     v4h);
  r.q[1] = __builtin_convertvector(*(const v4f*)(p0 + 4), v4h);
  r.q[2] = __builtin_convertvector(*(const v4f*)(p1),     v4h);
  r.q[3] = __builtin_convertvector(*(const v4f*)(p1 + 4), v4h);
  return r.v;
}

// Generic WMMA GEMM:  C[m][n] = sum_k A[m][k] * B(k,n),  K % 32 == 0 (guaranteed by caller)
//   BT=false: B stored [k][n], n contiguous (row stride LDB)   -> NN
//   BT=true : B stored [n][k], k contiguous (row stride LDB)   -> NT (Gram q.k^T)
// TM = M-subtiles per wave (M = gridDim.y * TM * 16 exactly). LDA/LDB compile-time.
// Batch: gz -> (b = gz/nh, h = gz%nh); pointer offsets b*s?b + h*s?h.
// Block = 128 threads = 4 waves; wave w owns columns n0 = bx*64 + w*16 and TM row tiles,
// reusing the B fragment TM times (B panels stay L2-resident, 192MB L2).
template<int TM, bool BT, int LDA, int LDB>
__global__ __launch_bounds__(128) void gemm_wmma(
    const float* __restrict__ A, const float* __restrict__ B, float* __restrict__ C,
    int N, int K, int ldc,
    long sAb, long sAh, long sBb, long sBh, long sCb, long sCh, int nh)
{
  const int wave = threadIdx.x >> 5;
  const int lane = threadIdx.x & 31;
  const int hi   = lane >> 4;      // 0: lanes 0-15, 1: lanes 16-31
  const int lcol = lane & 15;

  const int gz = blockIdx.z;
  const int bb = gz / nh, hh = gz % nh;
  const float* Ap = A + (long)bb * sAb + (long)hh * sAh;
  const float* Bp = B + (long)bb * sBb + (long)hh * sBh;
  float*       Cp = C + (long)bb * sCb + (long)hh * sCh;

  const int n0    = blockIdx.x * 64 + wave * 16;
  const int mbase = blockIdx.y * (TM * 16);
  if (n0 >= N) return;                       // wave-uniform: EXEC stays all-ones

  // Per-lane base pointers (strength-reduced; advanced each K-step)
  const float* arow[TM];
#pragma unroll
  for (int t = 0; t < TM; ++t)
    arow[t] = Ap + (long)(mbase + t * 16 + lcol) * LDA + hi * 16;   // contiguous run [0..15]
  const float* bptr = BT ? (Bp + (long)(n0 + lcol) * LDB + hi * 16)
                         : (Bp + n0 + lcol + (long)(hi * 16) * LDB);

  v8f acc[TM];
#pragma unroll
  for (int t = 0; t < TM; ++t)
#pragma unroll
    for (int v = 0; v < 8; ++v) acc[t][v] = 0.0f;

  for (int k0 = 0; k0 < K; k0 += 32) {
    // B fragment: lane holds col n0+lcol, half j -> k = k0 + hi*16 + j
    v16h bf;
    if (BT) {
      bf = cvt16(bptr, bptr + 8);
    } else {
      hfrag u;
#pragma unroll
      for (int j = 0; j < 16; ++j)           // 16 coalesced b32 loads, immediate offsets j*LDB*4
        u.e[j] = (_Float16)bptr[(long)j * LDB];
      bf = u.v;
    }
#pragma unroll
    for (int t = 0; t < TM; ++t) {
      v16h af = cvt16(arow[t], arow[t] + 8);
      acc[t] = __builtin_amdgcn_wmma_f32_16x16x32_f16(
          false, af, false, bf, (short)0, acc[t], false, false);
      arow[t] += 32;
    }
    bptr += BT ? 32L : 32L * LDB;
  }

#pragma unroll
  for (int t = 0; t < TM; ++t) {
    const int m0 = mbase + t * 16;
#pragma unroll
    for (int v = 0; v < 8; ++v)
      Cp[(long)(m0 + v + hi * 8) * ldc + n0 + lcol] = acc[t][v];
  }
}

// Depthwise 3x3, SAME padding, H=W=128. in/out: [b][Cch][128][128], w: [Cch][1][3][3]
__global__ __launch_bounds__(256) void dwconv3(
    const float* __restrict__ in, const float* __restrict__ w,
    float* __restrict__ out, int Cch, long total)
{
  long i = (long)blockIdx.x * 256 + threadIdx.x;
  if (i >= total) return;
  const int x = (int)(i & 127);
  const int y = (int)((i >> 7) & 127);
  const long cb = i >> 14;                 // b*Cch + c
  const int c = (int)(cb % Cch);
  const float* wp = w + c * 9;
  const float* p  = in + (cb << 14);
  float s = 0.0f;
#pragma unroll
  for (int dy = 0; dy < 3; ++dy) {
    int yy = y + dy - 1;
    if ((unsigned)yy < 128u) {
#pragma unroll
      for (int dx = 0; dx < 3; ++dx) {
        int xx = x + dx - 1;
        if ((unsigned)xx < 128u) s += p[yy * 128 + xx] * wp[dy * 3 + dx];
      }
    }
  }
  out[i] = s;
}

// L2 row norms of q and k rows of qkv buffer [b][576][hw].
// block r in [0,3072): r<1536 -> q row, else k row. rr = b*192 + c (c = head*48+d)
__global__ __launch_bounds__(256) void rownorm(
    const float* __restrict__ qkv, float* __restrict__ qn, float* __restrict__ kn)
{
  __shared__ float red[256];
  const int r = blockIdx.x;
  const bool isK = (r >= 1536);
  const int rr = isK ? r - 1536 : r;
  const int b = rr / 192, c = rr % 192;
  const float* src = qkv + (long)b * 576 * HWSZ + (long)((isK ? 192 : 0) + c) * HWSZ;
  float s = 0.0f;
  for (int i = threadIdx.x; i < (int)HWSZ; i += 256) { float v = src[i]; s += v * v; }
  red[threadIdx.x] = s; __syncthreads();
  for (int st = 128; st > 0; st >>= 1) {
    if (threadIdx.x < st) red[threadIdx.x] += red[threadIdx.x + st];
    __syncthreads();
  }
  if (threadIdx.x == 0) (isK ? kn : qn)[rr] = sqrtf(red[0]);
}

// Fold norms + temperature into raw Gram matrix, softmax over 48 cols (row stride 64).
// Cols 48..63 are zero-filled so the downstream attn@v GEMM can run with K=64 (K%32==0).
__global__ __launch_bounds__(64) void softmax48(
    float* __restrict__ G, const float* __restrict__ qn,
    const float* __restrict__ kn, const float* __restrict__ temp)
{
  __shared__ float red[64];
  const int r = blockIdx.x;                // b*192 + h*48 + d
  const int b = r / 192;
  const int h = (r % 192) / 48;
  float* row = G + (long)r * 64;
  const int e = threadIdx.x;
  const float t  = temp[h];
  const float qi = fmaxf(qn[r], EPSN);
  float val = -3.4e38f;
  if (e < 48) {
    float ki = fmaxf(kn[b * 192 + h * 48 + e], EPSN);
    val = row[e] / (qi * ki) * t;
  }
  red[e] = val; __syncthreads();
  for (int s = 32; s > 0; s >>= 1) { if (e < s) red[e] = fmaxf(red[e], red[e + s]); __syncthreads(); }
  const float mx = red[0]; __syncthreads();
  const float ex = (e < 48) ? expf(val - mx) : 0.0f;
  red[e] = ex; __syncthreads();
  for (int s = 32; s > 0; s >>= 1) { if (e < s) red[e] += red[e + s]; __syncthreads(); }
  row[e] = (e < 48) ? (ex / red[0]) : 0.0f;
}

extern "C" void kernel_launch(void* const* d_in, const int* in_sizes, int n_in,
                              void* d_out, int out_size, void* d_ws, size_t ws_size,
                              hipStream_t stream) {
  const float* x      = (const float*)d_in[0];
  const float* temp   = (const float*)d_in[1];
  const float* w_qkv  = (const float*)d_in[2];
  const float* w_dwq  = (const float*)d_in[3];
  const float* w_proj = (const float*)d_in[4];
  const float* w_kv   = (const float*)d_in[5];
  // d_in[6], d_in[7] = w_q1, w_q2 : dead FFT branch (DCE'd under jit)
  const float* w_dwkv = (const float*)d_in[8];
  const float* w_prjf = (const float*)d_in[9];
  float* out = (float*)d_out;

  // workspace carve-out
  float* R0 = (float*)d_ws;                 // [8][576][hw]  qkv pre-dw
  float* R1 = R0 + 8L * 576 * HWSZ;         // [8][576][hw]  qkv (q|k|v)
  float* R2 = R1 + 8L * 576 * HWSZ;         // [8][192][hw]  ping
  float* R3 = R2 + 8L * 192 * HWSZ;         // [8][192][hw]  pong
  float* G  = R3 + 8L * 192 * HWSZ;         // [1536][64]    Gram -> attn (cols 48..63 zero pad)
  float* qn = G  + 1536L * 64;
  float* kn = qn + 1536;

  const long sQKV = 576L * HWSZ, sC = 192L * HWSZ, sH = 48L * HWSZ;

  // 1) qkv = w_qkv @ x        [576x192]x[192x16384] per batch (M=576 -> gy=9, TM=4)
  gemm_wmma<4, false, 192, 16384><<<dim3(256, 9, 8), 128, 0, stream>>>(
      w_qkv, x, R0, 16384, 192, 16384, 0, 0, sC, 0, sQKV, 0, 1);
  // 2) depthwise 3x3 on 576 channels
  dwconv3<<<(8L * 576 * HWSZ + 255) / 256, 256, 0, stream>>>(R0, w_dwq, R1, 576, 8L * 576 * HWSZ);
  // 3) row L2 norms of q and k
  rownorm<<<3072, 256, 0, stream>>>(R1, qn, kn);
  // 4) Gram: G = q @ k^T  (raw; norms folded into softmax). M=48 -> TM=3, NT B = k rows
  gemm_wmma<3, true, 16384, 16384><<<dim3(1, 1, 32), 128, 0, stream>>>(
      R1, R1 + 192L * HWSZ, G, 48, 16384, 64,
      sQKV, sH, sQKV, sH, 4L * 48 * 64, 48L * 64, 4);
  // 5) attn = softmax(G / (||q|| ||k||) * temperature), zero-pad cols 48..63
  softmax48<<<1536, 64, 0, stream>>>(G, qn, kn, temp);
  // 6) out_att = attn @ v    [48x64]x[64x16384] per (b,h), pad K rows contribute 0
  gemm_wmma<3, false, 64, 16384><<<dim3(256, 1, 32), 128, 0, stream>>>(
      G, R1 + 384L * HWSZ, R2, 16384, 64, 16384,
      4L * 48 * 64, 48L * 64, sQKV, sH, sC, sH, 4);
  // 7) out = w_proj @ out_att   (M=192 -> gy=3, TM=4)
  gemm_wmma<4, false, 192, 16384><<<dim3(256, 3, 8), 128, 0, stream>>>(
      w_proj, R2, R3, 16384, 192, 16384, 0, 0, sC, 0, sC, 0, 1);
  // 8) vf_pre = w_kv[192:384] @ out   (kf half is dead)
  gemm_wmma<4, false, 192, 16384><<<dim3(256, 3, 8), 128, 0, stream>>>(
      w_kv + 192 * 192, R3, R2, 16384, 192, 16384, 0, 0, sC, 0, sC, 0, 1);
  // 9) vf = depthwise 3x3 (channels 192..383 of w_kv_dw)
  dwconv3<<<(8L * 192 * HWSZ + 255) / 256, 256, 0, stream>>>(R2, w_dwkv + 192 * 9, R3, 192, 8L * 192 * HWSZ);
  // 10) outf_att = attn @ vf
  gemm_wmma<3, false, 64, 16384><<<dim3(256, 1, 32), 128, 0, stream>>>(
      G, R3, R2, 16384, 64, 16384,
      4L * 48 * 64, 48L * 64, sC, sH, sC, sH, 4);
  // 11) outf = w_projf @ outf_att  -> d_out
  gemm_wmma<4, false, 192, 16384><<<dim3(256, 3, 8), 128, 0, stream>>>(
      w_prjf, R2, out, 16384, 192, 16384, 0, 0, sC, 0, sC, 0, 1);
}
